// SwinTransformerBlock_25632364822684
// MI455X (gfx1250) — compile-verified, matrix operand order checked by reading, not run on hardware
//
#include <hip/hip_runtime.h>
#include <cstddef>

// ---------------- constants ----------------
#define Bn   8
#define HR   128
#define WR   128
#define Cc   192
#define WSs  8
#define SSs  4
#define NHh  8
#define SPHh 6
#define CHHh 2
#define DHh  24
#define MIDm 48
#define LOGMAXC 4.605170185988091f
static const long Mtok = (long)Bn * HR * WR; // 131072

typedef __attribute__((ext_vector_type(16))) __bf16 v16bf;
typedef __attribute__((ext_vector_type(8)))  float  v8f;
typedef __attribute__((ext_vector_type(2)))  __bf16 bf2;

// ---------------- WMMA fragment loader (TN, K-major) ----------------
// A fragment: lane holds row M = lane%16; B fragment: lane holds col N = lane%16.
// 16 bf16 per lane cover K offsets: p<4 -> k0 + 8*(lane/16) + 2p ; p>=4 -> k0 + 16 + 8*(lane/16) + 2(p-4).
__device__ __forceinline__ v16bf frag_tn(const __bf16* __restrict__ src, long row, int ld,
                                         int k0, int K, int lane) {
  v16bf f;
  int hi = (lane >> 4) & 1;
  if (row >= 0 && (k0 + 32) <= K) {
    const bf2* pr = (const bf2*)(src + (size_t)row * ld + k0);
#pragma unroll
    for (int p = 0; p < 8; ++p) {
      int kk = (p < 4) ? (8 * hi + 2 * p) : (8 + 8 * hi + 2 * p);
      bf2 u = pr[kk >> 1];
      f[2 * p] = u.x; f[2 * p + 1] = u.y;
    }
  } else {
#pragma unroll
    for (int p = 0; p < 8; ++p) {
      int kk = k0 + ((p < 4) ? (8 * hi + 2 * p) : (8 + 8 * hi + 2 * p));
      f[2 * p]     = (row >= 0 && kk     < K) ? src[(size_t)row * ld + kk]     : (__bf16)0.0f;
      f[2 * p + 1] = (row >= 0 && kk + 1 < K) ? src[(size_t)row * ld + kk + 1] : (__bf16)0.0f;
    }
  }
  return f;
}

// ---------------- generic TN GEMM: out[m,n] = act(sum_k A[m,k]*W[n,k] + bias[n]) ----------------
// ACT: 0 none, 1 relu, 2 exact gelu.  convTap>=0: gather A rows with 3x3 spatial offset (zero pad).
// NT: N-tiles per wave (A fragment reused across NT WMMAs). Requires (N/16) % NT == 0.
template <int ACT, bool BIAS, bool ACCUM, bool RESID, int NT>
__global__ __launch_bounds__(256) void gemm_tn(const __bf16* __restrict__ A,
                                               const __bf16* __restrict__ W,
                                               const float* __restrict__ bias,
                                               float* outF, __bf16* outB,
                                               const float* __restrict__ resid,
                                               int M, int N, int K, int convTap) {
  int lane = threadIdx.x & 31;
  int wave = threadIdx.x >> 5;
  int ntn_g = (N >> 4) / NT;                 // tile-groups along N
  long tile = (long)blockIdx.x * 8 + wave;
  long ntiles = (long)(M >> 4) * ntn_g;
  if (tile >= ntiles) return;
  int tm = (int)(tile / ntn_g);
  int tn0 = (int)(tile % ntn_g) * NT;
  int r = lane & 15;
  int hi = lane >> 4;
  long arow = (long)tm * 16 + r;
  if (convTap >= 0) {
    long t = arow;
    int xx = (int)(t & 127); t >>= 7;
    int yy = (int)(t & 127); t >>= 7;
    int bb = (int)t;
    int dy = convTap / 3 - 1, dx = convTap % 3 - 1;
    int ys = yy + dy, xs = xx + dx;
    arow = (ys < 0 || ys > 127 || xs < 0 || xs > 127) ? -1
         : (((long)bb * 128 + ys) * 128 + xs);
  }
  v8f acc[NT] = {};
  for (int k0 = 0; k0 < K; k0 += 32) {
    v16bf a = frag_tn(A, arow, K, k0, K, lane);   // loaded once, reused NT times
#pragma unroll
    for (int nt = 0; nt < NT; ++nt) {
      long wrow = (long)(tn0 + nt) * 16 + r;
      v16bf b = frag_tn(W, wrow, K, k0, K, lane);
      acc[nt] = __builtin_amdgcn_wmma_f32_16x16x32_bf16(false, a, false, b, (short)0,
                                                        acc[nt], false, false);
    }
  }
#pragma unroll
  for (int nt = 0; nt < NT; ++nt) {
    int n = (tn0 + nt) * 16 + r;
    float bv = BIAS ? bias[n] : 0.0f;
#pragma unroll
    for (int v = 0; v < 8; ++v) {
      long m = (long)tm * 16 + hi * 8 + v;
      long idx = m * N + n;
      float val = acc[nt][v] + bv;
      if (ACT == 1) val = fmaxf(val, 0.0f);
      else if (ACT == 2) val = 0.5f * val * (1.0f + erff(val * 0.70710678118f));
      if (RESID) val += resid[idx];
      if (ACCUM) val += outF[idx];
      if (outF) outF[idx] = val;
      if (outB) outB[idx] = (__bf16)val;
    }
  }
}

// ---------------- small elementwise / reduction kernels ----------------
__global__ void convert_bf16(const float* __restrict__ s, __bf16* __restrict__ d, long n) {
  long i = (long)blockIdx.x * 256 + threadIdx.x;
  if (i < n) d[i] = (__bf16)s[i];
}

// torch conv weight (O,I,3,3) -> tap-major bf16 [9][O][I]
__global__ void repack_conv(const float* __restrict__ w, __bf16* __restrict__ dst, int O, int I) {
  long idx = (long)blockIdx.x * 256 + threadIdx.x;
  long total = (long)O * I * 9;
  if (idx >= total) return;
  int tap = (int)(idx / ((long)O * I));
  long rest = idx % ((long)O * I);
  int o = (int)(rest / I), i = (int)(rest % I);
  dst[idx] = (__bf16)w[((long)o * I + i) * 9 + tap];
}

// LayerNorm over C=192, one wave per token, bf16 out
__global__ __launch_bounds__(256) void ln1k(const float* __restrict__ x,
                                            const float* __restrict__ g,
                                            const float* __restrict__ b,
                                            __bf16* __restrict__ h) {
  long tok = (long)blockIdx.x * 8 + (threadIdx.x >> 5);
  int lane = threadIdx.x & 31;
  float vals[6]; float s = 0.0f;
#pragma unroll
  for (int i = 0; i < 6; ++i) { vals[i] = x[tok * Cc + lane + i * 32]; s += vals[i]; }
  for (int m = 16; m; m >>= 1) s += __shfl_xor(s, m, 32);
  float mean = s * (1.0f / Cc);
  float vs = 0.0f;
#pragma unroll
  for (int i = 0; i < 6; ++i) { float d = vals[i] - mean; vs += d * d; }
  for (int m = 16; m; m >>= 1) vs += __shfl_xor(vs, m, 32);
  float inv = rsqrtf(vs * (1.0f / Cc) + 1e-5f);
#pragma unroll
  for (int i = 0; i < 6; ++i) {
    int c = lane + i * 32;
    h[tok * Cc + c] = (__bf16)((vals[i] - mean) * inv * g[c] + b[c]);
  }
}

__device__ __forceinline__ int swin_region(int y) { return y < 120 ? 0 : (y < 124 ? 1 : 2); }

// one block = (b, window, head): shifted-window cosine attention, WMMA QK^T and P*V
__global__ __launch_bounds__(256) void win_attn(const __bf16* __restrict__ qkv,
                                                const float* __restrict__ logit_sp,
                                                const float* __restrict__ rpb,
                                                __bf16* __restrict__ outSp) {
  __shared__ __bf16 qn[64][32];
  __shared__ __bf16 kn[64][32];
  __shared__ __bf16 vT[32][64];
  __shared__ float  att[64][64];
  __shared__ __bf16 pb[64][64];
  int id = blockIdx.x;
  int head = id % 6; id /= 6;
  int win = id % 256; id /= 256;
  int b = id;
  int wy = win >> 4, wx = win & 15;
  int tid = threadIdx.x;
  float scale = __expf(fminf(logit_sp[head], LOGMAXC));
  if (tid < 64) {
    int py = tid >> 3, px = tid & 7;
    int y = (wy * 8 + py + SSs) & 127;
    int x = (wx * 8 + px + SSs) & 127;
    const __bf16* src = qkv + (((long)(b * 128 + y) * 128 + x) * 576 + head * 72);
    float q[24], k[24]; float qs = 0.0f, ks = 0.0f;
#pragma unroll
    for (int d = 0; d < 24; ++d) { q[d] = (float)src[d];      qs += q[d] * q[d]; }
#pragma unroll
    for (int d = 0; d < 24; ++d) { k[d] = (float)src[24 + d]; ks += k[d] * k[d]; }
    float qr = scale / fmaxf(sqrtf(qs), 1e-12f);
    float kr = 1.0f  / fmaxf(sqrtf(ks), 1e-12f);
#pragma unroll
    for (int d = 0; d < 24; ++d) { qn[tid][d] = (__bf16)(q[d] * qr); kn[tid][d] = (__bf16)(k[d] * kr); }
#pragma unroll
    for (int d = 24; d < 32; ++d) { qn[tid][d] = (__bf16)0.0f; kn[tid][d] = (__bf16)0.0f; }
#pragma unroll
    for (int d = 0; d < 24; ++d) vT[d][tid] = src[48 + d];
    if (tid < 8) { for (int j = 0; j < 64; ++j) vT[24 + tid][j] = (__bf16)0.0f; }
  }
  __syncthreads();
  int lane = tid & 31, wv = tid >> 5;
  int r = lane & 15, hi = lane >> 4;
  // QK^T : 16 tiles of 16x16, 2 per wave
  for (int t = wv * 2; t < wv * 2 + 2; ++t) {
    int ti = t >> 2, tj = t & 3;
    v16bf a  = frag_tn(&qn[0][0], ti * 16 + r, 32, 0, 32, lane);
    v16bf bb = frag_tn(&kn[0][0], tj * 16 + r, 32, 0, 32, lane);
    v8f acc = {};
    acc = __builtin_amdgcn_wmma_f32_16x16x32_bf16(false, a, false, bb, (short)0, acc, false, false);
#pragma unroll
    for (int v = 0; v < 8; ++v) {
      int i = ti * 16 + hi * 8 + v;
      int j = tj * 16 + r;
      int iy = i >> 3, ix = i & 7, jy = j >> 3, jx = j & 7;
      int ridx = (iy - jy + 7) * 15 + (ix - jx + 7);
      int li = swin_region(wy * 8 + iy) * 3 + swin_region(wx * 8 + ix);
      int lj = swin_region(wy * 8 + jy) * 3 + swin_region(wx * 8 + jx);
      float msk = (li != lj) ? -100.0f : 0.0f;
      att[i][j] = acc[v] + rpb[ridx * SPHh + head] + msk;
    }
  }
  __syncthreads();
  if (tid < 64) {
    float mx = -1e30f;
    for (int j = 0; j < 64; ++j) mx = fmaxf(mx, att[tid][j]);
    float s = 0.0f;
    for (int j = 0; j < 64; ++j) s += __expf(att[tid][j] - mx);
    float inv = 1.0f / s;
    for (int j = 0; j < 64; ++j) pb[tid][j] = (__bf16)(__expf(att[tid][j] - mx) * inv);
  }
  __syncthreads();
  // P * V : 4x2 tiles, one per wave
  {
    int tiM = wv >> 1, tiN = wv & 1;
    v8f acc = {};
    for (int k0 = 0; k0 < 64; k0 += 32) {
      v16bf a  = frag_tn(&pb[0][0], tiM * 16 + r, 64, k0, 64, lane);
      v16bf bb = frag_tn(&vT[0][0], tiN * 16 + r, 64, k0, 64, lane);
      acc = __builtin_amdgcn_wmma_f32_16x16x32_bf16(false, a, false, bb, (short)0, acc, false, false);
    }
    int d = tiN * 16 + r;
    if (d < 24) {
#pragma unroll
      for (int v = 0; v < 8; ++v) {
        int i = tiM * 16 + hi * 8 + v;
        int ys = wy * 8 + (i >> 3), xs = wx * 8 + (i & 7);
        long row = ((long)(b * 128 + ys) * 128 + xs);
        outSp[row * 144 + head * 24 + d] = (__bf16)acc[v];
      }
    }
  }
}

__global__ void roll_copy144(const __bf16* __restrict__ src, __bf16* __restrict__ dst) {
  long i = (long)blockIdx.x * 256 + threadIdx.x;
  if (i >= Mtok * 144) return;
  int c = (int)(i % 144); long t = i / 144;
  int x = (int)(t % 128); long t2 = t / 128;
  int y = (int)(t2 % 128); int b = (int)(t2 / 128);
  int ys = (y - SSs) & 127, xs = (x - SSs) & 127;
  dst[i] = src[(((long)b * 128 + ys) * 128 + xs) * 144 + c];
}

// channel attention: row norms of qc/kc (768 rows of 16384)
__global__ void ch_norms(const __bf16* __restrict__ qkv, float* __restrict__ norms) {
  __shared__ float red[256];
  int idx = blockIdx.x;
  int c = idx % 24; int h = (idx / 24) % 2; int b = (idx / 48) % 8; int qk = idx / 384;
  int col = (SPHh + h) * 72 + qk * 24 + c;
  const __bf16* base = qkv + (long)b * 16384 * 576;
  float s = 0.0f;
  for (int i = threadIdx.x; i < 16384; i += 256) {
    float v = (float)base[(long)i * 576 + col]; s += v * v;
  }
  red[threadIdx.x] = s; __syncthreads();
  for (int st = 128; st; st >>= 1) { if (threadIdx.x < st) red[threadIdx.x] += red[threadIdx.x + st]; __syncthreads(); }
  if (threadIdx.x == 0) norms[idx] = fmaxf(sqrtf(red[0]), 1e-12f);
}

__global__ void ch_dots(const __bf16* __restrict__ qkv, float* __restrict__ araw) {
  __shared__ float red[256];
  int idx = blockIdx.x;
  int d = idx % 24; int c = (idx / 24) % 24; int h = (idx / 576) % 2; int b = idx / 1152;
  int colq = (SPHh + h) * 72 + c;
  int colk = (SPHh + h) * 72 + 24 + d;
  const __bf16* base = qkv + (long)b * 16384 * 576;
  float s = 0.0f;
  for (int i = threadIdx.x; i < 16384; i += 256)
    s += (float)base[(long)i * 576 + colq] * (float)base[(long)i * 576 + colk];
  red[threadIdx.x] = s; __syncthreads();
  for (int st = 128; st; st >>= 1) { if (threadIdx.x < st) red[threadIdx.x] += red[threadIdx.x + st]; __syncthreads(); }
  if (threadIdx.x == 0) araw[idx] = red[0];
}

__global__ void ch_softmax(const float* __restrict__ araw, const float* __restrict__ norms,
                           const float* __restrict__ logit_ch, float* __restrict__ afin) {
  int t = blockIdx.x * blockDim.x + threadIdx.x;
  if (t >= 384) return;
  int c = t % 24; int h = (t / 24) % 2; int b = t / 48;
  float scale = __expf(fminf(logit_ch[h], LOGMAXC));
  float nq = norms[b * 48 + h * 24 + c];
  float a[24]; float mx = -1e30f;
  for (int d = 0; d < 24; ++d) {
    float nk = norms[384 + b * 48 + h * 24 + d];
    float v = araw[((long)(b * 2 + h) * 576) + c * 24 + d] / (nq * nk) * scale;
    a[d] = v; mx = fmaxf(mx, v);
  }
  float s = 0.0f;
  for (int d = 0; d < 24; ++d) { a[d] = __expf(a[d] - mx); s += a[d]; }
  float inv = 1.0f / s;
  for (int d = 0; d < 24; ++d) afin[((long)(b * 2 + h) * 576) + c * 24 + d] = a[d] * inv;
}

__global__ __launch_bounds__(256) void ch_apply(const __bf16* __restrict__ qkv,
                                                const float* __restrict__ afin,
                                                __bf16* __restrict__ oc) {
  __shared__ float a2[1152];
  int b = blockIdx.x / 64;  // 256 tokens/block, 16384/256=64 blocks per image
  for (int i = threadIdx.x; i < 1152; i += 256) a2[i] = afin[(long)b * 1152 + i];
  __syncthreads();
  long t = (long)blockIdx.x * 256 + threadIdx.x;
  for (int h = 0; h < 2; ++h) {
    float vv[24];
#pragma unroll
    for (int d = 0; d < 24; ++d) vv[d] = (float)qkv[t * 576 + (SPHh + h) * 72 + 48 + d];
    for (int c = 0; c < 24; ++c) {
      float s = 0.0f;
#pragma unroll
      for (int d = 0; d < 24; ++d) s += a2[h * 576 + c * 24 + d] * vv[d];
      oc[t * 48 + h * 24 + c] = (__bf16)s;
    }
  }
}

__global__ void colmean(const float* __restrict__ src, float* __restrict__ out) {
  __shared__ float red[256];
  int c = blockIdx.x % Cc; int b = blockIdx.x / Cc;
  float s = 0.0f;
  for (int i = threadIdx.x; i < 16384; i += 256) s += src[((long)b * 16384 + i) * Cc + c];
  red[threadIdx.x] = s; __syncthreads();
  for (int st = 128; st; st >>= 1) { if (threadIdx.x < st) red[threadIdx.x] += red[threadIdx.x + st]; __syncthreads(); }
  if (threadIdx.x == 0) out[blockIdx.x] = red[0] * (1.0f / 16384.0f);
}

__global__ void gate_mlp(const float* __restrict__ mean, const float* __restrict__ w1,
                         const float* __restrict__ w2, const float* __restrict__ b2,
                         float* __restrict__ gate) {
  __shared__ float hid[48];
  __shared__ float mv[192];
  int b = blockIdx.x;
  for (int i = threadIdx.x; i < 192; i += 256) mv[i] = mean[b * 192 + i];
  __syncthreads();
  if (threadIdx.x < 48) {
    float s = 0.0f;
    for (int c = 0; c < 192; ++c) s += mv[c] * w1[threadIdx.x * 192 + c];
    hid[threadIdx.x] = fmaxf(s, 0.0f);
  }
  __syncthreads();
  if (threadIdx.x < 192) {
    float s = b2[threadIdx.x];
    for (int m = 0; m < 48; ++m) s += hid[m] * w2[threadIdx.x * 48 + m];
    gate[b * 192 + threadIdx.x] = 1.0f / (1.0f + __expf(-s));
  }
}

__global__ void cfca(const float* __restrict__ sp, const float* __restrict__ ch,
                     const float* __restrict__ s2c, const float* __restrict__ c2s,
                     const float* __restrict__ gs, const float* __restrict__ gc,
                     __bf16* __restrict__ cat) {
  long i = (long)blockIdx.x * 256 + threadIdx.x;
  if (i >= Mtok * Cc) return;
  int c = (int)(i % Cc); long t = i / Cc; int b = (int)(t / 16384);
  float spf = sp[i] + c2s[i] * gc[b * Cc + c];
  float chf = ch[i] + s2c[i] * gs[b * Cc + c];
  cat[t * (2 * Cc) + c] = (__bf16)spf;
  cat[t * (2 * Cc) + Cc + c] = (__bf16)chf;
}

__global__ void bnrelu(const float* __restrict__ in, const float* __restrict__ g,
                       const float* __restrict__ bb, const float* __restrict__ mn,
                       const float* __restrict__ vr, __bf16* outB, float* outF,
                       int N, long total) {
  long i = (long)blockIdx.x * 256 + threadIdx.x;
  if (i >= total) return;
  int c = (int)(i % N);
  float val = (in[i] - mn[c]) * g[c] * rsqrtf(vr[c] + 1e-5f) + bb[c];
  val = fmaxf(val, 0.0f);
  if (outB) outB[i] = (__bf16)val;
  if (outF) outF[i] = val;
}

__global__ void relu_to_bf16(const float* __restrict__ in, __bf16* __restrict__ out, long n) {
  long i = (long)blockIdx.x * 256 + threadIdx.x;
  if (i < n) out[i] = (__bf16)fmaxf(in[i], 0.0f);
}

__global__ void sggate(const __bf16* __restrict__ sg, const float* __restrict__ w2,
                       const float* __restrict__ b2, float* __restrict__ g) {
  long t = (long)blockIdx.x * 256 + threadIdx.x;
  if (t >= Mtok) return;
  float s = b2[0];
#pragma unroll 8
  for (int m = 0; m < 96; ++m) s += (float)sg[t * 96 + m] * w2[m];
  g[t] = 1.0f / (1.0f + __expf(-s));
}

// x2 = x + f*(1+g); h2 = LN2(x2)  (one wave per token)
__global__ __launch_bounds__(256) void fuse_ln2(const float* __restrict__ x,
                                                const float* __restrict__ f,
                                                const float* __restrict__ g,
                                                const float* __restrict__ ln2g,
                                                const float* __restrict__ ln2b,
                                                float* __restrict__ x2,
                                                __bf16* __restrict__ h2) {
  long tok = (long)blockIdx.x * 8 + (threadIdx.x >> 5);
  int lane = threadIdx.x & 31;
  float gv = g[tok] + 1.0f;
  float vals[6]; float s = 0.0f;
#pragma unroll
  for (int i = 0; i < 6; ++i) {
    long idx = tok * Cc + lane + i * 32;
    float v = x[idx] + f[idx] * gv;
    vals[i] = v; s += v;
  }
  for (int m = 16; m; m >>= 1) s += __shfl_xor(s, m, 32);
  float mean = s * (1.0f / Cc);
  float vs = 0.0f;
#pragma unroll
  for (int i = 0; i < 6; ++i) { float d = vals[i] - mean; vs += d * d; }
  for (int m = 16; m; m >>= 1) vs += __shfl_xor(vs, m, 32);
  float inv = rsqrtf(vs * (1.0f / Cc) + 1e-5f);
#pragma unroll
  for (int i = 0; i < 6; ++i) {
    int c = lane + i * 32;
    long idx = tok * Cc + c;
    x2[idx] = vals[i];
    h2[idx] = (__bf16)((vals[i] - mean) * inv * ln2g[c] + ln2b[c]);
  }
}

// ---------------- host ----------------
static inline int gblocks(long M, long N, int NT) {
  return (int)(((M / 16) * ((N / 16) / NT) + 7) / 8);
}

extern "C" void kernel_launch(void* const* d_in, const int* in_sizes, int n_in,
                              void* d_out, int out_size, void* d_ws, size_t ws_size,
                              hipStream_t stream) {
  (void)in_sizes; (void)n_in; (void)out_size; (void)ws_size;
  const float* x        = (const float*)d_in[0];
  const float* ln1_g    = (const float*)d_in[1];
  const float* ln1_b    = (const float*)d_in[2];
  const float* w_qkv    = (const float*)d_in[3];
  const float* b_qkv    = (const float*)d_in[4];
  const float* logit_sp = (const float*)d_in[5];
  const float* rpb      = (const float*)d_in[6];
  const float* w_spp    = (const float*)d_in[7];
  const float* b_spp    = (const float*)d_in[8];
  const float* logit_ch = (const float*)d_in[9];
  const float* w_chp    = (const float*)d_in[10];
  const float* b_chp    = (const float*)d_in[11];
  const float* w_spproj = (const float*)d_in[12];
  const float* w_chproj = (const float*)d_in[13];
  const float* w_s2c1   = (const float*)d_in[14];
  const float* w_s2c2   = (const float*)d_in[15];
  const float* b_s2c2   = (const float*)d_in[16];
  const float* w_c2s1   = (const float*)d_in[17];
  const float* w_c2s2   = (const float*)d_in[18];
  const float* b_c2s2   = (const float*)d_in[19];
  const float* w_gs1    = (const float*)d_in[20];
  const float* w_gs2    = (const float*)d_in[21];
  const float* b_gs2    = (const float*)d_in[22];
  const float* w_gc1    = (const float*)d_in[23];
  const float* w_gc2    = (const float*)d_in[24];
  const float* b_gc2    = (const float*)d_in[25];
  const float* w_f1     = (const float*)d_in[26];
  const float* bn1_g    = (const float*)d_in[27];
  const float* bn1_b    = (const float*)d_in[28];
  const float* bn1_m    = (const float*)d_in[29];
  const float* bn1_v    = (const float*)d_in[30];
  const float* w_f2     = (const float*)d_in[31];
  const float* bn2_g    = (const float*)d_in[32];
  const float* bn2_b    = (const float*)d_in[33];
  const float* bn2_m    = (const float*)d_in[34];
  const float* bn2_v    = (const float*)d_in[35];
  const float* w_sg1    = (const float*)d_in[36];
  const float* w_sg2    = (const float*)d_in[37];
  const float* b_sg2    = (const float*)d_in[38];
  const float* ln2_g    = (const float*)d_in[39];
  const float* ln2_b    = (const float*)d_in[40];
  const float* w_fc1    = (const float*)d_in[41];
  const float* b_fc1    = (const float*)d_in[42];
  const float* w_fc2    = (const float*)d_in[43];
  const float* b_fc2    = (const float*)d_in[44];

  char* ws = (char*)d_ws;
  size_t off = 0;
  auto alloc = [&](size_t bytes) { void* p = ws + off; off += (bytes + 255) & ~(size_t)255; return p; };
  const long M = Mtok;

  __bf16* h_bf    = (__bf16*)alloc(M * 192 * 2);   // reused for h2
  __bf16* qkv_bf  = (__bf16*)alloc(M * 576 * 2);
  __bf16* attn_o  = (__bf16*)alloc(M * 144 * 2);   // reused for rolled
  __bf16* spp_o   = (__bf16*)alloc(M * 144 * 2);
  float*  spF     = (float*)alloc(M * 192 * 4);    // reused for f1raw
  __bf16* spB     = (__bf16*)alloc(M * 192 * 2);   // reused for f0_bf
  float*  chF     = (float*)alloc(M * 192 * 4);    // reused for convacc
  __bf16* chB     = (__bf16*)alloc(M * 192 * 2);   // reused for f_bf
  __bf16* oc      = (__bf16*)alloc(M * 48 * 2);
  __bf16* chp_o   = (__bf16*)alloc(M * 48 * 2);
  __bf16* s2ch    = (__bf16*)alloc(M * 48 * 2);
  __bf16* c2sh    = (__bf16*)alloc(M * 48 * 2);
  float*  s2c     = (float*)alloc(M * 192 * 4);    // reused for fF
  float*  c2s     = (float*)alloc(M * 192 * 4);    // reused for x2
  __bf16* cat     = (__bf16*)alloc(M * 384 * 2);
  float*  sgacc   = (float*)alloc(M * 96 * 4);
  __bf16* sg_bf   = (__bf16*)alloc(M * 96 * 2);
  float*  gsc     = (float*)alloc(M * 4);
  __bf16* fc1o    = (__bf16*)alloc(M * 768 * 2);
  float*  norms   = (float*)alloc(768 * 4);
  float*  araw    = (float*)alloc(9216 * 4);
  float*  afin    = (float*)alloc(16 * 576 * 4);
  float*  mean_s  = (float*)alloc(Bn * 192 * 4);
  float*  mean_c  = (float*)alloc(Bn * 192 * 4);
  float*  gate_s  = (float*)alloc(Bn * 192 * 4);
  float*  gate_c  = (float*)alloc(Bn * 192 * 4);
  __bf16* bqkv    = (__bf16*)alloc(576 * 192 * 2);
  __bf16* bspp    = (__bf16*)alloc(144 * 144 * 2);
  __bf16* bchp    = (__bf16*)alloc(48 * 48 * 2);
  __bf16* bspproj = (__bf16*)alloc(192 * 144 * 2);
  __bf16* bchproj = (__bf16*)alloc(192 * 48 * 2);
  __bf16* bs2c1   = (__bf16*)alloc(48 * 192 * 2);
  __bf16* bs2c2   = (__bf16*)alloc(192 * 48 * 2);
  __bf16* bc2s1   = (__bf16*)alloc(48 * 192 * 2);
  __bf16* bc2s2   = (__bf16*)alloc(192 * 48 * 2);
  __bf16* bf1     = (__bf16*)alloc(192 * 384 * 2);
  __bf16* btapf2  = (__bf16*)alloc(9L * 192 * 192 * 2);
  __bf16* btapsg  = (__bf16*)alloc(9L * 96 * 192 * 2);
  __bf16* bfc1    = (__bf16*)alloc(768 * 192 * 2);
  __bf16* bfc2    = (__bf16*)alloc(192 * 768 * 2);

  auto cvt = [&](const float* s, __bf16* d, long n) {
    convert_bf16<<<(int)((n + 255) / 256), 256, 0, stream>>>(s, d, n);
  };
  cvt(w_qkv, bqkv, 576 * 192);     cvt(w_spp, bspp, 144 * 144);
  cvt(w_chp, bchp, 48 * 48);       cvt(w_spproj, bspproj, 192 * 144);
  cvt(w_chproj, bchproj, 192 * 48);cvt(w_s2c1, bs2c1, 48 * 192);
  cvt(w_s2c2, bs2c2, 192 * 48);    cvt(w_c2s1, bc2s1, 48 * 192);
  cvt(w_c2s2, bc2s2, 192 * 48);    cvt(w_f1, bf1, 192 * 384);
  cvt(w_fc1, bfc1, 768 * 192);     cvt(w_fc2, bfc2, 192 * 768);
  repack_conv<<<(int)((9L * 192 * 192 + 255) / 256), 256, 0, stream>>>(w_f2, btapf2, 192, 192);
  repack_conv<<<(int)((9L * 96 * 192 + 255) / 256), 256, 0, stream>>>(w_sg1, btapsg, 96, 192);

  // LN1 -> h (bf16)
  ln1k<<<(int)(M / 8), 256, 0, stream>>>(x, ln1_g, ln1_b, h_bf);
  // qkv = h @ w_qkv.T + b   (N=576, NT=4)
  gemm_tn<0, true, false, false, 4><<<gblocks(M, 576, 4), 256, 0, stream>>>(
      h_bf, bqkv, b_qkv, nullptr, qkv_bf, nullptr, (int)M, 576, 192, -1);
  // shifted-window attention -> attn_o (shifted coords, bf16 [M,144])
  win_attn<<<Bn * 256 * 6, 256, 0, stream>>>(qkv_bf, logit_sp, rpb, attn_o);
  // o @ w_spp.T + b_spp   (N=144, NT=3)
  gemm_tn<0, true, false, false, 3><<<gblocks(M, 144, 3), 256, 0, stream>>>(
      attn_o, bspp, b_spp, nullptr, spp_o, nullptr, (int)M, 144, 144, -1);
  // roll back (+SS,+SS) into attn_o buffer, then @ w_sp_proj.T -> sp_out (f32 + bf16)
  roll_copy144<<<(int)((M * 144 + 255) / 256), 256, 0, stream>>>(spp_o, attn_o);
  gemm_tn<0, false, false, false, 4><<<gblocks(M, 192, 4), 256, 0, stream>>>(
      attn_o, bspproj, nullptr, spF, spB, nullptr, (int)M, 192, 144, -1);
  // channel attention
  ch_norms<<<768, 256, 0, stream>>>(qkv_bf, norms);
  ch_dots<<<9216, 256, 0, stream>>>(qkv_bf, araw);
  ch_softmax<<<2, 192, 0, stream>>>(araw, norms, logit_ch, afin);
  ch_apply<<<(int)(M / 256), 256, 0, stream>>>(qkv_bf, afin, oc);
  gemm_tn<0, true, false, false, 3><<<gblocks(M, 48, 3), 256, 0, stream>>>(
      oc, bchp, b_chp, nullptr, chp_o, nullptr, (int)M, 48, 48, -1);
  gemm_tn<0, false, false, false, 4><<<gblocks(M, 192, 4), 256, 0, stream>>>(
      chp_o, bchproj, nullptr, chF, chB, nullptr, (int)M, 192, 48, -1);
  // CFCA gates
  colmean<<<Bn * 192, 256, 0, stream>>>(spF, mean_s);
  colmean<<<Bn * 192, 256, 0, stream>>>(chF, mean_c);
  gate_mlp<<<Bn, 256, 0, stream>>>(mean_s, w_gs1, w_gs2, b_gs2, gate_s);
  gate_mlp<<<Bn, 256, 0, stream>>>(mean_c, w_gc1, w_gc2, b_gc2, gate_c);
  // cross MLPs
  gemm_tn<1, false, false, false, 3><<<gblocks(M, 48, 3), 256, 0, stream>>>(
      spB, bs2c1, nullptr, nullptr, s2ch, nullptr, (int)M, 48, 192, -1);
  gemm_tn<0, true, false, false, 4><<<gblocks(M, 192, 4), 256, 0, stream>>>(
      s2ch, bs2c2, b_s2c2, s2c, nullptr, nullptr, (int)M, 192, 48, -1);
  gemm_tn<1, false, false, false, 3><<<gblocks(M, 48, 3), 256, 0, stream>>>(
      chB, bc2s1, nullptr, nullptr, c2sh, nullptr, (int)M, 48, 192, -1);
  gemm_tn<0, true, false, false, 4><<<gblocks(M, 192, 4), 256, 0, stream>>>(
      c2sh, bc2s2, b_c2s2, c2s, nullptr, nullptr, (int)M, 192, 48, -1);
  cfca<<<(int)((M * 192 + 255) / 256), 256, 0, stream>>>(spF, chF, s2c, c2s, gate_s, gate_c, cat);
  // XFF fusion: f1 GEMM -> BN1+relu -> conv3x3 (9 WMMA taps) -> BN2+relu
  float* f1raw = spF; __bf16* f0_bf = spB; float* convacc = chF;
  float* fF = s2c;    __bf16* f_bf = chB;  float* x2 = c2s;  __bf16* h2 = h_bf;
  gemm_tn<0, false, false, false, 4><<<gblocks(M, 192, 4), 256, 0, stream>>>(
      cat, bf1, nullptr, f1raw, nullptr, nullptr, (int)M, 192, 384, -1);
  bnrelu<<<(int)((M * 192 + 255) / 256), 256, 0, stream>>>(
      f1raw, bn1_g, bn1_b, bn1_m, bn1_v, f0_bf, nullptr, 192, M * 192);
  for (int tap = 0; tap < 9; ++tap) {
    const __bf16* wt = btapf2 + (long)tap * 192 * 192;
    if (tap == 0)
      gemm_tn<0, false, false, false, 4><<<gblocks(M, 192, 4), 256, 0, stream>>>(
          f0_bf, wt, nullptr, convacc, nullptr, nullptr, (int)M, 192, 192, tap);
    else
      gemm_tn<0, false, true, false, 4><<<gblocks(M, 192, 4), 256, 0, stream>>>(
          f0_bf, wt, nullptr, convacc, nullptr, nullptr, (int)M, 192, 192, tap);
  }
  bnrelu<<<(int)((M * 192 + 255) / 256), 256, 0, stream>>>(
      convacc, bn2_g, bn2_b, bn2_m, bn2_v, f_bf, fF, 192, M * 192);
  // spatial gate: conv3x3(f) -> relu -> dot w_sg2 -> sigmoid
  for (int tap = 0; tap < 9; ++tap) {
    const __bf16* wt = btapsg + (long)tap * 96 * 192;
    if (tap == 0)
      gemm_tn<0, false, false, false, 3><<<gblocks(M, 96, 3), 256, 0, stream>>>(
          f_bf, wt, nullptr, sgacc, nullptr, nullptr, (int)M, 96, 192, tap);
    else
      gemm_tn<0, false, true, false, 3><<<gblocks(M, 96, 3), 256, 0, stream>>>(
          f_bf, wt, nullptr, sgacc, nullptr, nullptr, (int)M, 96, 192, tap);
  }
  relu_to_bf16<<<(int)((M * 96 + 255) / 256), 256, 0, stream>>>(sgacc, sg_bf, M * 96);
  sggate<<<(int)((M + 255) / 256), 256, 0, stream>>>(sg_bf, w_sg2, b_sg2, gsc);
  // residual + LN2, then MLP
  fuse_ln2<<<(int)(M / 8), 256, 0, stream>>>(x, fF, gsc, ln2_g, ln2_b, x2, h2);
  gemm_tn<2, true, false, false, 4><<<gblocks(M, 768, 4), 256, 0, stream>>>(
      h2, bfc1, b_fc1, nullptr, fc1o, nullptr, (int)M, 768, 192, -1);
  gemm_tn<0, true, false, true, 4><<<gblocks(M, 192, 4), 256, 0, stream>>>(
      fc1o, bfc2, b_fc2, (float*)d_out, nullptr, x2, (int)M, 192, 768, -1);
}
